// MultiFeatureDecoder_19756849562197
// MI455X (gfx1250) — matrix operand order repaired and checked by reading, not compiled, not run on hardware
//
#include <hip/hip_runtime.h>
#include <hip/hip_bf16.h>
#include <cstdint>
#include <cstddef>

// MultiFeatureDecoder for MI455X (gfx1250, wave32, WMMA).
// B=64, N=1024, D=512, C=512, K=16.

typedef __attribute__((ext_vector_type(16))) _Float16 v16h;
typedef __attribute__((ext_vector_type(8)))  _Float16 v8h;
typedef __attribute__((ext_vector_type(4)))  _Float16 v4h;
typedef __attribute__((ext_vector_type(8)))  float    v8f;

#define B_    64
#define N_    1024
#define D_    512
#define C_    512
#define TOPK  16
#define GAMMA 0.3f
#define INV_TEMP (1.0f/0.07f)
#define EPS_  1e-12f

// Padded A-staging stride: 520 halfs = 260 dwords == 4 (mod 64 banks)
// -> conflict-minimal ds_load_b128 A-fragment reads in the GEMM hot loop.
#define AS_   520

// CDNA5 async global->LDS staging (ASYNCcnt path). Set to 0 to fall back to
// plain vectorized loads+ds_stores if the assembler rejects the mnemonics.
#define USE_ASYNC_LDS 1

// ---------------------------------------------------------------------------
// Kernel 1: normalize text rows; write f32 [C,D] and f16 transposed [D,C].
// ---------------------------------------------------------------------------
__global__ __launch_bounds__(256) void prep_text(const float* __restrict__ text,
                                                 float* __restrict__ text_norm,
                                                 _Float16* __restrict__ text_t) {
    __shared__ float red[256];
    const int c = blockIdx.x;
    const int t = threadIdx.x;
    const float* row = text + (size_t)c * D_;
    float x0 = row[t];
    float x1 = row[t + 256];
    red[t] = x0 * x0 + x1 * x1;
    __syncthreads();
    for (int s = 128; s > 0; s >>= 1) {
        if (t < s) red[t] += red[t + s];
        __syncthreads();
    }
    const float inv = 1.0f / fmaxf(sqrtf(red[0]), EPS_);
    const float y0 = x0 * inv, y1 = x1 * inv;
    text_norm[(size_t)c * D_ + t]       = y0;
    text_norm[(size_t)c * D_ + t + 256] = y1;
    // transposed f16 copy: text_t[d][c]  (B-matrix operand for WMMA)
    text_t[(size_t)t * C_ + c]         = (_Float16)y0;
    text_t[(size_t)(t + 256) * C_ + c] = (_Float16)y1;
}

// ---------------------------------------------------------------------------
// Kernel 2: global branch: aff_g[b][c] = softmax_c( (cls_n . text_n) / TEMP )
// ---------------------------------------------------------------------------
__global__ __launch_bounds__(256) void global_aff(const float* __restrict__ v_cls,
                                                  const float* __restrict__ text_norm,
                                                  float* __restrict__ aff_g) {
    __shared__ float red[256];
    __shared__ float clsrow[D_];
    __shared__ float logit[C_];
    const int b = blockIdx.x;
    const int t = threadIdx.x;
    float x0 = v_cls[(size_t)b * D_ + t];
    float x1 = v_cls[(size_t)b * D_ + t + 256];
    red[t] = x0 * x0 + x1 * x1;
    __syncthreads();
    for (int s = 128; s > 0; s >>= 1) {
        if (t < s) red[t] += red[t + s];
        __syncthreads();
    }
    const float inv = 1.0f / fmaxf(sqrtf(red[0]), EPS_);
    clsrow[t]       = x0 * inv;
    clsrow[t + 256] = x1 * inv;
    __syncthreads();

    for (int cc = 0; cc < 2; ++cc) {
        const int c = t + cc * 256;
        const float* tr = text_norm + (size_t)c * D_;
        float acc = 0.0f;
        #pragma unroll 4
        for (int d = 0; d < D_; d += 4) {
            acc += clsrow[d]     * tr[d];
            acc += clsrow[d + 1] * tr[d + 1];
            acc += clsrow[d + 2] * tr[d + 2];
            acc += clsrow[d + 3] * tr[d + 3];
        }
        logit[c] = acc * INV_TEMP;
    }
    __syncthreads();

    // block softmax over 512
    red[t] = fmaxf(logit[t], logit[t + 256]);
    __syncthreads();
    for (int s = 128; s > 0; s >>= 1) {
        if (t < s) red[t] = fmaxf(red[t], red[t + s]);
        __syncthreads();
    }
    const float m = red[0];
    __syncthreads();
    const float e0 = __expf(logit[t] - m);
    const float e1 = __expf(logit[t + 256] - m);
    red[t] = e0 + e1;
    __syncthreads();
    for (int s = 128; s > 0; s >>= 1) {
        if (t < s) red[t] += red[t + s];
        __syncthreads();
    }
    const float invsum = 1.0f / red[0];
    aff_g[(size_t)b * C_ + t]       = e0 * invsum;
    aff_g[(size_t)b * C_ + t + 256] = e1 * invsum;
}

// ---------------------------------------------------------------------------
// Kernel 3: main WMMA kernel. One block = 16 patch rows x full C (512).
//   - async-stage 16x512 f32 patches into LDS (global_load_async_to_lds_b128)
//   - row inv-norms (fold 1/TEMP), convert to padded f16 A staging
//   - 8 waves x 4 col-tiles each, K in steps of 32 via v_wmma_f32_16x16x32_f16
//   - fused row softmax in LDS, b128 store of aff_l [B*N, C] f32
// ---------------------------------------------------------------------------
__global__ __launch_bounds__(256) void patch_kernel(const float* __restrict__ patches,
                                                    const _Float16* __restrict__ text_t,
                                                    float* __restrict__ aff_l) {
    __shared__ __align__(32) _Float16 ldsA[16 * AS_];  // ~16.6 KB padded f16 A
    __shared__ __align__(16) float    ldsL[16 * C_];   // 32 KB: f32 staging -> logits
    __shared__ float rowred[16 * 16];
    __shared__ float rowa[16];   // inv-norm * 1/TEMP, later row max
    __shared__ float rowb[16];   // 1 / row sum of exp

    const int t    = threadIdx.x;
    const int row0 = blockIdx.x * 16;           // first patch row (flat over B*N)
    const float4* src4 = (const float4*)(patches + (size_t)row0 * D_);
    float4* ldsL4 = (float4*)ldsL;

    // Phase 1: stage 16x512 f32 (2048 float4, 8 per thread, coalesced)
#if USE_ASYNC_LDS
    {
        const uint32_t lbase = (uint32_t)(uintptr_t)ldsL;   // LDS byte offset
        #pragma unroll
        for (int i = 0; i < 8; ++i) {
            const int e4 = t + 256 * i;
            const uint32_t laddr = lbase + (uint32_t)e4 * 16u;
            const float4* ga = src4 + e4;
            asm volatile("global_load_async_to_lds_b128 %0, %1, off"
                         :: "v"(laddr), "v"(ga) : "memory");
        }
        asm volatile("s_wait_asynccnt 0x0" ::: "memory");
    }
#else
    #pragma unroll
    for (int i = 0; i < 8; ++i) ldsL4[t + 256 * i] = src4[t + 256 * i];
#endif
    __syncthreads();

    // Phase 2: per-row sum of squares -> inv-norm (fold 1/TEMP)
    {
        const int r = t >> 4, l = t & 15;
        float s = 0.0f;
        #pragma unroll
        for (int j = 0; j < 32; ++j) {
            const float v = ldsL[r * C_ + l + 16 * j];
            s += v * v;
        }
        rowred[r * 16 + l] = s;
    }
    __syncthreads();
    if (t < 16) {
        float s = 0.0f;
        #pragma unroll
        for (int j = 0; j < 16; ++j) s += rowred[t * 16 + j];
        rowa[t] = (1.0f / fmaxf(sqrtf(s), EPS_)) * INV_TEMP;
    }
    __syncthreads();

    // Phase 3: convert to padded f16 A staging (8-byte LDS stores)
    #pragma unroll
    for (int i = 0; i < 8; ++i) {
        const int e4   = t + 256 * i;
        const int r    = e4 >> 7;          // 128 float4 per row
        const int col4 = e4 & 127;
        const float4 lf = ldsL4[e4];
        const float  sc = rowa[r];
        v4h h;
        h.x = (_Float16)(lf.x * sc);
        h.y = (_Float16)(lf.y * sc);
        h.z = (_Float16)(lf.z * sc);
        h.w = (_Float16)(lf.w * sc);
        *(v4h*)&ldsA[r * AS_ + col4 * 4] = h;
    }
    __syncthreads();

    // Phase 4: GEMM. wave w owns column tiles [w*64, w*64+64).
    const int lane  = t & 31;
    const int wv    = t >> 5;
    const int r16   = lane & 15;          // A row in the 16-row tile
    const int kbase = (lane >> 4) * 8;    // A K sub-offset per ISA 16-bit layout
    const int cb    = wv * 64;            // wave's first output column

    v8f acc[4] = {};
    union AU { v16h v; v8h h[2]; };

    for (int k0 = 0; k0 < D_; k0 += 32) {
        // A fragment: lanes 0-15 hold K k0+0..7 / k0+16..23 of row r16,
        //             lanes 16-31 hold K k0+8..15 / k0+24..31.
        AU au;
        au.h[0] = *(const v8h*)&ldsA[r16 * AS_ + k0 + kbase];
        au.h[1] = *(const v8h*)&ldsA[r16 * AS_ + k0 + 16 + kbase];
        const v16h a = au.v;

        const _Float16* bp = text_t + (size_t)(k0 + kbase) * C_ + cb + r16;
        if (k0 + 32 < D_) {
            __builtin_prefetch((const void*)(text_t + (size_t)(k0 + 32 + lane) * C_ + cb), 0, 1);
        }
        #pragma unroll
        for (int j = 0; j < 4; ++j) {
            // B fragment: lane l holds column N=l%16, K in the same order as A.
            v16h bb;
            #pragma unroll
            for (int s = 0; s < 8; ++s) {
                bb[s]     = bp[(size_t)s * C_ + j * 16];
                bb[s + 8] = bp[(size_t)(16 + s) * C_ + j * 16];
            }
            acc[j] = __builtin_amdgcn_wmma_f32_16x16x32_f16(
                false, a, false, bb, (short)0, acc[j], false, false);
        }
    }

    // Phase 5: scatter accumulators (logits, already /TEMP) into ldsL.
    // C/D layout: VGPR v, lanes 0-15 -> M=v, lanes 16-31 -> M=v+8; N = lane&15.
    {
        const int colw   = cb + r16;
        const int rowoff = (lane >> 4) * 8;
        #pragma unroll
        for (int j = 0; j < 4; ++j) {
            #pragma unroll
            for (int v = 0; v < 8; ++v) {
                ldsL[(v + rowoff) * C_ + colw + j * 16] = acc[j][v];
            }
        }
    }
    __syncthreads();

    // Phase 6: row softmax over C=512
    {
        const int r = t >> 4, l = t & 15;
        float m = -1e30f;
        #pragma unroll
        for (int j = 0; j < 32; ++j) m = fmaxf(m, ldsL[r * C_ + l + 16 * j]);
        rowred[r * 16 + l] = m;
    }
    __syncthreads();
    if (t < 16) {
        float m = rowred[t * 16];
        #pragma unroll
        for (int j = 1; j < 16; ++j) m = fmaxf(m, rowred[t * 16 + j]);
        rowa[t] = m;
    }
    __syncthreads();
    {
        const int r = t >> 4, l = t & 15;
        const float m = rowa[r];
        float s = 0.0f;
        #pragma unroll
        for (int j = 0; j < 32; ++j) {
            const int idx = r * C_ + l + 16 * j;
            const float e = __expf(ldsL[idx] - m);
            ldsL[idx] = e;
            s += e;
        }
        rowred[r * 16 + l] = s;
    }
    __syncthreads();
    if (t < 16) {
        float s = 0.0f;
        #pragma unroll
        for (int j = 0; j < 16; ++j) s += rowred[t * 16 + j];
        rowb[t] = 1.0f / s;
    }
    __syncthreads();

    // Phase 7: b128 store of normalized probabilities
    float4* dst4 = (float4*)(aff_l + (size_t)row0 * C_);
    #pragma unroll
    for (int i = 0; i < 8; ++i) {
        const int e4 = t + 256 * i;
        const int r  = e4 >> 7;
        const float s = rowb[r];
        float4 v = ldsL4[e4];
        v.x *= s; v.y *= s; v.z *= s; v.w *= s;
        dst4[e4] = v;
    }
}

// ---------------------------------------------------------------------------
// Kernel 4: per (b,c) top-16 over N=1024, mean, combine with aff_g.
// Static-index register top-k (no scratch spills). Coalesced over c.
// ---------------------------------------------------------------------------
__global__ __launch_bounds__(256) void topk_kernel(const float* __restrict__ aff_l,
                                                   const float* __restrict__ aff_g,
                                                   float* __restrict__ out) {
    const int b = blockIdx.x >> 1;
    const int c = ((blockIdx.x & 1) << 8) + threadIdx.x;
    const float* col = aff_l + (size_t)b * N_ * C_ + c;

    float tk[TOPK];
    #pragma unroll
    for (int i = 0; i < TOPK; ++i) tk[i] = -1.0f;   // softmax outputs are >= 0
    float cmin = -1.0f;

    for (int n = 0; n < N_; ++n) {
        const float v = col[(size_t)n * C_];
        if (v > cmin) {
            // replace first slot equal to current min (static indexing only)
            bool done = false;
            #pragma unroll
            for (int i = 0; i < TOPK; ++i) {
                const bool hit = (!done) && (tk[i] == cmin);
                if (hit) { tk[i] = v; done = true; }
            }
            float m = tk[0];
            #pragma unroll
            for (int i = 1; i < TOPK; ++i) m = fminf(m, tk[i]);
            cmin = m;
        }
    }
    float s = 0.0f;
    #pragma unroll
    for (int i = 0; i < TOPK; ++i) s += tk[i];
    const size_t o = (size_t)b * C_ + c;
    out[o] = GAMMA * aff_g[o] + (1.0f - GAMMA) * (s * (1.0f / (float)TOPK));
}

// ---------------------------------------------------------------------------
// Launch. Workspace layout (bytes):
//   [0,            524288)  : text_t   f16 [D][C]
//   [524288,      1572864)  : text_norm f32 [C][D]
//   [1572864,     1703936)  : aff_g    f32 [B][C]
//   [1703936,   135921664)  : aff_l    f32 [B*N][C]   (~128 MB)
// ---------------------------------------------------------------------------
extern "C" void kernel_launch(void* const* d_in, const int* in_sizes, int n_in,
                              void* d_out, int out_size, void* d_ws, size_t ws_size,
                              hipStream_t stream) {
    const float* v_cls     = (const float*)d_in[0];   // [B, D]
    const float* v_patches = (const float*)d_in[1];   // [B, N, D]
    const float* text      = (const float*)d_in[2];   // [C, D]
    float* out = (float*)d_out;                       // [B, C]

    uint8_t* ws = (uint8_t*)d_ws;
    _Float16* text_t    = (_Float16*)(ws);
    float*    text_norm = (float*)(ws + 524288);
    float*    aff_g     = (float*)(ws + 524288 + 1048576);
    float*    aff_l     = (float*)(ws + 524288 + 1048576 + 131072);

    prep_text   <<<C_,             256, 0, stream>>>(text, text_norm, text_t);
    global_aff  <<<B_,             256, 0, stream>>>(v_cls, text_norm, aff_g);
    patch_kernel<<<(B_ * N_) / 16, 256, 0, stream>>>(v_patches, text_t, aff_l);
    topk_kernel <<<B_ * 2,         256, 0, stream>>>(aff_l, aff_g, out);
}
